// LinearAttention_20358144983287
// MI455X (gfx1250) — compile-verified, matrix operand order checked by reading, not run on hardware
//
#include <hip/hip_runtime.h>
#include <hip/hip_bf16.h>
#include <cstdint>
#include <cstddef>

// ---------------------------------------------------------------------------
// CDNA5 WMMA types (wave32): bf16 A/B fragments (16 x bf16), f32 accum (8 x f32)
// ---------------------------------------------------------------------------
typedef __bf16 bf16_t;
typedef bf16_t v16bf __attribute__((ext_vector_type(16)));
typedef float  v8f   __attribute__((ext_vector_type(8)));
typedef int    v4i32 __attribute__((ext_vector_type(4)));

typedef __attribute__((address_space(1))) v4i32* glb_v4i_p;
typedef __attribute__((address_space(3))) v4i32* lds_v4i_p;

union FragU { v16bf v; uint4 q[2]; };

// Feature detection: async global->LDS DMA (gfx1250)
#if defined(__has_builtin)
#  if __has_builtin(__builtin_amdgcn_global_load_async_to_lds_b128)
#    define USE_ASYNC_LDS 1
#  endif
#endif
#ifndef USE_ASYNC_LDS
#  define USE_ASYNC_LDS 0
#endif

__device__ __forceinline__ void wait_async0() {
#if defined(__has_builtin) && __has_builtin(__builtin_amdgcn_s_wait_asynccnt)
    __builtin_amdgcn_s_wait_asynccnt(0);
#else
    asm volatile("s_wait_asynccnt 0x0" ::: "memory");
#endif
}

// Problem constants (fixed by setup_inputs)
static constexpr int BB   = 4;
static constexpr int TT   = 4096;
static constexpr int HID  = 1024;
static constexpr int HH   = 16;
static constexpr int DD   = 64;
static constexpr int MM   = BB * TT;        // 16384 rows
static constexpr int CHK  = 128;            // scan chunk length
static constexpr int NCHK = TT / CHK;       // 32 chunks

__device__ __forceinline__ unsigned short f32_to_bf16(float f) {
    unsigned int u = __float_as_uint(f);
    u += 0x7FFFu + ((u >> 16) & 1u);        // round-to-nearest-even
    return (unsigned short)(u >> 16);
}

__device__ __forceinline__ float bf16_to_f32(unsigned short h) {
    return __uint_as_float(((unsigned)h) << 16);
}

__device__ __forceinline__ float phi_fn(float x) {   // elu(x)+1
    return x > 0.f ? x + 1.f : __expf(x);
}

__device__ __forceinline__ float sigmoid_fn(float x) {
    return 1.f / (1.f + __expf(-x));
}

// ---------------------------------------------------------------------------
// fp32 -> bf16 bulk convert (one float4 -> 4 bf16 per thread)
// ---------------------------------------------------------------------------
__global__ __launch_bounds__(256)
void cvt_f32_to_bf16(const float* __restrict__ src, unsigned short* __restrict__ dst) {
    size_t i = (size_t)blockIdx.x * 256 + threadIdx.x;
    float4 v = *((const float4*)src + i);
    uint2 p;
    p.x = (unsigned)f32_to_bf16(v.x) | ((unsigned)f32_to_bf16(v.y) << 16);
    p.y = (unsigned)f32_to_bf16(v.z) | ((unsigned)f32_to_bf16(v.w) << 16);
    *((uint2*)dst + i) = p;
}

// ---------------------------------------------------------------------------
// Fused bf16-WMMA GEMM:  C[M,N] = A[M,K] * Bw[N,K]^T  (bf16 in, f32 accum)
// Block tile 128x128 (8 waves, wave tile 32x64 -> 8 WMMA/k-step), BK=32,
// double-buffered LDS (pad 40 halves: conflict-free ds_load_b128).
// Staging: GLOBAL_LOAD_ASYNC_TO_LDS_B128 (ASYNCcnt) when available --
// zero VGPR residency, no scratch spill; else register staging fallback.
// EPI 0: store fp32 C -> d_out
// EPI 1: qkv: phi(q), phi(k), v scattered fp32 to [B,H,T,D]
// EPI 2: gate: g=sigmoid(acc+bgate); mid_bf16 = g*attn + (1-g)*v_direct
// ---------------------------------------------------------------------------
template <int EPI>
__global__ __launch_bounds__(256, 1)
void gemm_bf16_wmma(const unsigned short* __restrict__ A,
                    const unsigned short* __restrict__ Bw,
                    int M, int N, int K,
                    float* __restrict__ Cout,
                    float* __restrict__ qArr, float* __restrict__ kArr,
                    float* __restrict__ vArr,
                    const float* __restrict__ bgate,
                    unsigned short* __restrict__ midBf) {
    constexpr int LDA  = 40;                 // padded LDS stride (halves)
    constexpr int TILE = 128 * LDA;          // 5120 halves per 128x32 tile
    __shared__ unsigned short lds[4 * TILE]; // [A0 | B0 | A1 | B1] = 40 KB

    const int tid   = threadIdx.x;
    const int wave  = tid >> 5;
    const int lane  = tid & 31;
    const int lr    = lane & 15;             // row/col within 16
    const int hs    = lane >> 4;             // half-select (K-chunk parity)
    const int waveM = (wave & 3) * 32;       // 4 waves along M
    const int waveN = (wave >> 2) * 64;      // 2 waves along N

    const int blockM = blockIdx.x * 128;
    const int blockN = blockIdx.y * 128;

    v8f acc[2][4];
    const v8f vzero = {0.f, 0.f, 0.f, 0.f, 0.f, 0.f, 0.f, 0.f};
#pragma unroll
    for (int i = 0; i < 2; ++i)
#pragma unroll
        for (int j = 0; j < 4; ++j) acc[i][j] = vzero;

    // shared compute: aF persistent, bF transient (low register pressure)
    auto compute = [&](int buf) {
        const unsigned short* sA = lds + buf * 2 * TILE;
        const unsigned short* sB = sA + TILE;
        FragU aF[2];
#pragma unroll
        for (int ti = 0; ti < 2; ++ti) {
            const unsigned short* p = &sA[(waveM + ti * 16 + lr) * LDA];
            aF[ti].q[0] = *(const uint4*)(p + 8 * hs);       // K 0-7  / 8-15
            aF[ti].q[1] = *(const uint4*)(p + 16 + 8 * hs);  // K16-23 / 24-31
        }
#pragma unroll
        for (int tj = 0; tj < 4; ++tj) {
            FragU bF;
            const unsigned short* p = &sB[(waveN + tj * 16 + lr) * LDA];
            bF.q[0] = *(const uint4*)(p + 8 * hs);
            bF.q[1] = *(const uint4*)(p + 16 + 8 * hs);
            acc[0][tj] = __builtin_amdgcn_wmma_f32_16x16x32_bf16(
                false, aF[0].v, false, bF.v, (short)0, acc[0][tj], false, false);
            acc[1][tj] = __builtin_amdgcn_wmma_f32_16x16x32_bf16(
                false, aF[1].v, false, bF.v, (short)0, acc[1][tj], false, false);
        }
    };

    const int kTiles = K >> 5;

#if USE_ASYNC_LDS
    // ---------- async global -> LDS staging (no VGPR residency) ----------
    auto issue_tile = [&](int kk, int buf) {
        unsigned short* dA = lds + buf * 2 * TILE;
        unsigned short* dB = dA + TILE;
#pragma unroll
        for (int i = 0; i < 2; ++i) {
            int fi  = tid + i * 256;         // 0..511 16B-chunk slots
            int row = fi >> 2;
            int ch  = fi & 3;
            const unsigned short* pA = A  + (size_t)(blockM + row) * K + kk + ch * 8;
            const unsigned short* pB = Bw + (size_t)(blockN + row) * K + kk + ch * 8;
            __builtin_amdgcn_global_load_async_to_lds_b128(
                (glb_v4i_p)(uintptr_t)pA,
                (lds_v4i_p)&dA[row * LDA + ch * 8],
                0, 0);
            __builtin_amdgcn_global_load_async_to_lds_b128(
                (glb_v4i_p)(uintptr_t)pB,
                (lds_v4i_p)&dB[row * LDA + ch * 8],
                0, 0);
        }
    };

    issue_tile(0, 0);
    for (int kt = 0; kt < kTiles; ++kt) {
        wait_async0();                       // my tile-kt DMAs landed in LDS
        __syncthreads();                     // all waves visible; buf^1 free
        const int cur = kt & 1;
        if (kt + 1 < kTiles)
            issue_tile((kt + 1) << 5, cur ^ 1);  // DMA overlaps WMMA
        compute(cur);
    }
#else
    // ---------- fallback: register staging ----------
    uint4 ra[2], rb[2];
    auto load_regs = [&](int kk) {
#pragma unroll
        for (int i = 0; i < 2; ++i) {
            int fi  = tid + i * 256;
            int row = fi >> 2;
            int ch  = fi & 3;
            const unsigned short* pA = A  + (size_t)(blockM + row) * K + kk + ch * 8;
            const unsigned short* pB = Bw + (size_t)(blockN + row) * K + kk + ch * 8;
            ra[i] = *(const uint4*)pA;
            rb[i] = *(const uint4*)pB;
            __builtin_prefetch(pA + 32, 0, 0);
            __builtin_prefetch(pB + 32, 0, 0);
        }
    };
    auto store_lds = [&](int buf) {
        unsigned short* dA = lds + buf * 2 * TILE;
        unsigned short* dB = dA + TILE;
#pragma unroll
        for (int i = 0; i < 2; ++i) {
            int fi  = tid + i * 256;
            int row = fi >> 2;
            int ch  = fi & 3;
            *(uint4*)&dA[row * LDA + ch * 8] = ra[i];
            *(uint4*)&dB[row * LDA + ch * 8] = rb[i];
        }
    };
    load_regs(0);
    store_lds(0);
    for (int kt = 0; kt < kTiles; ++kt) {
        __syncthreads();
        const int cur = kt & 1;
        const bool more = (kt + 1 < kTiles);
        if (more) load_regs((kt + 1) << 5);
        compute(cur);
        if (more) store_lds(cur ^ 1);
    }
#endif

    // ---- epilogue: C layout VGPR r -> (M = r + 8*hs, N = lr) ----
#pragma unroll
    for (int ti = 0; ti < 2; ++ti)
#pragma unroll
        for (int tj = 0; tj < 4; ++tj)
#pragma unroll
            for (int r = 0; r < 8; ++r) {
                int gm = blockM + waveM + ti * 16 + r + 8 * hs;
                int gn = blockN + waveN + tj * 16 + lr;
                float val = acc[ti][tj][r];
                if constexpr (EPI == 0) {
                    Cout[(size_t)gm * N + gn] = val;
                } else if constexpr (EPI == 1) {
                    int b = gm >> 12, t = gm & (TT - 1);
                    if (gn < HID) {
                        int h = gn >> 6, d = gn & 63;
                        qArr[(((size_t)(b * HH + h)) * TT + t) * DD + d] = phi_fn(val);
                    } else if (gn < 2 * HID) {
                        int c = gn - HID; int h = c >> 6, d = c & 63;
                        kArr[(((size_t)(b * HH + h)) * TT + t) * DD + d] = phi_fn(val);
                    } else {
                        int c = gn - 2 * HID; int h = c >> 6, d = c & 63;
                        vArr[(((size_t)(b * HH + h)) * TT + t) * DD + d] = val;
                    }
                } else { // EPI == 2 : gate (A operand IS attn_bf16, N == K)
                    float g  = sigmoid_fn(val + bgate[gn]);
                    float av = bf16_to_f32(A[(size_t)gm * K + gn]);
                    int b = gm >> 12, t = gm & (TT - 1);
                    int h = gn >> 6, d = gn & 63;
                    float vd = vArr[(((size_t)(b * HH + h)) * TT + t) * DD + d];
                    midBf[(size_t)gm * N + gn] = f32_to_bf16(g * av + (1.f - g) * vd);
                }
            }
}

// ---------------------------------------------------------------------------
// Scan phase 1: per-chunk aggregates with zero init.  grid = B*H*NCHK x 64
// ---------------------------------------------------------------------------
__global__ __launch_bounds__(64)
void scan_chunk_agg(const float* __restrict__ kA, const float* __restrict__ vA,
                    const float* __restrict__ decay_param,
                    float* __restrict__ aggKV, float* __restrict__ aggK) {
    int unit = blockIdx.x;
    int d = threadIdx.x;
    int c = unit % NCHK;
    int h = (unit / NCHK) % HH;
    int b = unit / (NCHK * HH);
    float decay = sigmoid_fn(decay_param[h]);
    size_t base = (((size_t)(b * HH + h)) * TT + (size_t)c * CHK) * DD + d;
    float skv = 0.f, sk = 0.f;
    for (int i = 0; i < CHK; ++i) {
        float k = kA[base + (size_t)i * DD];
        float v = vA[base + (size_t)i * DD];
        skv = decay * skv + k * v;
        sk  = decay * sk + k;
    }
    size_t o = ((size_t)(b * HH + h) * NCHK + c) * DD + d;
    aggKV[o] = skv;
    aggK[o]  = sk;
}

// ---------------------------------------------------------------------------
// Scan phase 2: sequential combine of chunk aggregates -> per-chunk init state.
// ---------------------------------------------------------------------------
__global__ __launch_bounds__(256)
void scan_combine(const float* __restrict__ aggKV, const float* __restrict__ aggK,
                  const float* __restrict__ decay_param,
                  float* __restrict__ iniKV, float* __restrict__ iniK) {
    int id = blockIdx.x * blockDim.x + threadIdx.x;    // < B*H*D
    int d = id % DD;
    int h = (id / DD) % HH;
    int b = id / (DD * HH);
    float decay = sigmoid_fn(decay_param[h]);
    float dC = __powf(decay, (float)CHK);
    float skv = 0.f, sk = 0.f;
    for (int c = 0; c < NCHK; ++c) {
        size_t o = ((size_t)(b * HH + h) * NCHK + c) * DD + d;
        iniKV[o] = skv;
        iniK[o]  = sk;
        skv = dC * skv + aggKV[o];
        sk  = dC * sk + aggK[o];
    }
}

// ---------------------------------------------------------------------------
// Scan phase 3: in-chunk finalize + normalization; writes bf16 attn [B,T,HID].
// wave32 = one (b,h,chunk); lane owns d=lane and d=lane+32; shfl_xor reduce.
// ---------------------------------------------------------------------------
__global__ __launch_bounds__(256)
void scan_final(const float* __restrict__ qA, const float* __restrict__ kA,
                const float* __restrict__ vA,
                const float* __restrict__ iniKV, const float* __restrict__ iniK,
                const float* __restrict__ decay_param,
                unsigned short* __restrict__ attnBf) {
    int unit = blockIdx.x * 8 + (threadIdx.x >> 5);    // < B*H*NCHK = 2048
    int lane = threadIdx.x & 31;
    int c = unit % NCHK;
    int h = (unit / NCHK) % HH;
    int b = unit / (NCHK * HH);
    float decay = sigmoid_fn(decay_param[h]);

    size_t io = ((size_t)(b * HH + h) * NCHK + c) * DD;
    float skv0 = iniKV[io + lane], skv1 = iniKV[io + lane + 32];
    float sk0  = iniK[io + lane],  sk1  = iniK[io + lane + 32];

    size_t base = (((size_t)(b * HH + h)) * TT + (size_t)c * CHK) * DD;
    for (int i = 0; i < CHK; ++i) {
        size_t p = base + (size_t)i * DD;
        float q0 = qA[p + lane], q1 = qA[p + lane + 32];
        float k0 = kA[p + lane], k1 = kA[p + lane + 32];
        float v0 = vA[p + lane], v1 = vA[p + lane + 32];
        skv0 = decay * skv0 + k0 * v0;
        skv1 = decay * skv1 + k1 * v1;
        sk0  = decay * sk0 + k0;
        sk1  = decay * sk1 + k1;
        float part = q0 * sk0 + q1 * sk1;
#pragma unroll
        for (int off = 16; off > 0; off >>= 1)
            part += __shfl_xor(part, off, 32);
        float den = fmaxf(part, 1e-6f);
        int t = c * CHK + i;
        size_t ob = ((size_t)b * TT + t) * HID + h * DD;
        attnBf[ob + lane]      = f32_to_bf16(q0 * skv0 / den);
        attnBf[ob + lane + 32] = f32_to_bf16(q1 * skv1 / den);
    }
}

// ---------------------------------------------------------------------------
// Host-side orchestration (graph-capture safe: launches only)
// ---------------------------------------------------------------------------
extern "C" void kernel_launch(void* const* d_in, const int* in_sizes, int n_in,
                              void* d_out, int out_size, void* d_ws, size_t ws_size,
                              hipStream_t stream) {
    const float* x           = (const float*)d_in[0];
    const float* Wqkv        = (const float*)d_in[1];
    const float* Wout        = (const float*)d_in[2];
    const float* Wgate       = (const float*)d_in[3];
    const float* bgate       = (const float*)d_in[4];
    const float* decay_param = (const float*)d_in[5];

    const int M = MM, K = HID;
    const size_t n = (size_t)M * HID;               // 16.78M elements

    char* p = (char*)d_ws;
    float* qA = (float*)p;              p += n * sizeof(float);
    float* kA = (float*)p;              p += n * sizeof(float);
    float* vA = (float*)p;              p += n * sizeof(float);
    unsigned short* xBf    = (unsigned short*)p;  p += n * 2;
    unsigned short* attnBf = (unsigned short*)p;  p += n * 2;
    unsigned short* wqkvBf = (unsigned short*)p;  p += (size_t)3 * HID * HID * 2;
    unsigned short* wgBf   = (unsigned short*)p;  p += (size_t)HID * HID * 2;
    unsigned short* woBf   = (unsigned short*)p;  p += (size_t)HID * HID * 2;
    const size_t na = (size_t)BB * HH * NCHK * DD;
    float* aggKV = (float*)p;           p += na * sizeof(float);
    float* aggK  = (float*)p;           p += na * sizeof(float);
    float* iniKV = (float*)p;           p += na * sizeof(float);
    float* iniK  = (float*)p;           p += na * sizeof(float);
    unsigned short* midBf = xBf;        // alias: x dead after qkv GEMM

    dim3 blk(256);

    // 0) one-shot bf16 conversion of streamed operands
    cvt_f32_to_bf16<<<(unsigned)(n / 1024), blk, 0, stream>>>(x, xBf);
    cvt_f32_to_bf16<<<(unsigned)((size_t)3 * HID * HID / 1024), blk, 0, stream>>>(Wqkv, wqkvBf);
    cvt_f32_to_bf16<<<(unsigned)((size_t)HID * HID / 1024), blk, 0, stream>>>(Wgate, wgBf);
    cvt_f32_to_bf16<<<(unsigned)((size_t)HID * HID / 1024), blk, 0, stream>>>(Wout, woBf);

    // 1) qkv projection + phi, scatter to [B,H,T,D]
    gemm_bf16_wmma<1><<<dim3(M / 128, (3 * HID) / 128), blk, 0, stream>>>(
        xBf, wqkvBf, M, 3 * HID, K, nullptr, qA, kA, vA, nullptr, nullptr);

    // 2) chunked linear-recurrence scan (3 phases) -> bf16 attn
    scan_chunk_agg<<<BB * HH * NCHK, 64, 0, stream>>>(kA, vA, decay_param, aggKV, aggK);
    scan_combine<<<(BB * HH * DD) / 256, 256, 0, stream>>>(aggKV, aggK, decay_param, iniKV, iniK);
    scan_final<<<(BB * HH * NCHK) / 8, 256, 0, stream>>>(qA, kA, vA, iniKV, iniK,
                                                         decay_param, attnBf);

    // 3) gate projection + sigmoid mix -> bf16 mid (aliases xBf)
    gemm_bf16_wmma<2><<<dim3(M / 128, HID / 128), blk, 0, stream>>>(
        attnBf, wgBf, M, HID, K, nullptr, nullptr, nullptr, vA, bgate, midBf);

    // 4) output projection -> d_out (fp32)
    gemm_bf16_wmma<0><<<dim3(M / 128, HID / 128), blk, 0, stream>>>(
        midBf, woBf, M, HID, K, (float*)d_out, nullptr, nullptr, nullptr,
        nullptr, nullptr);
}
